// GraphModule_59012850647684
// MI455X (gfx1250) — matrix operand order repaired and checked by reading, not compiled, not run on hardware
//
#include <hip/hip_runtime.h>
#include <stdint.h>

// ---------- vector types (avoid HIP class-wrapped vectors in unions) ----------
typedef __bf16        v16bf __attribute__((ext_vector_type(16)));
typedef float         v8f   __attribute__((ext_vector_type(8)));
typedef float         f32x4 __attribute__((ext_vector_type(4)));
typedef unsigned int  u32x4 __attribute__((ext_vector_type(4)));
typedef unsigned int  u32x2 __attribute__((ext_vector_type(2)));

union Frag { u32x4 q[2]; v16bf v; };

#define FDIM 128

// ---------- helpers ----------
__device__ __forceinline__ unsigned int f2bf(float a) {
  unsigned int u = __float_as_uint(a);
  u = (u + 0x7FFFu + ((u >> 16) & 1u)) >> 16;
  return u;
}
__device__ __forceinline__ unsigned int pk_bf16(float a, float b) {
  return f2bf(a) | (f2bf(b) << 16);
}

__device__ __forceinline__ v8f wmma_bf16(v16bf a, v16bf b, v8f c) {
  return __builtin_amdgcn_wmma_f32_16x16x32_bf16(false, a, false, b, (short)0, c, false, false);
}

// Native CDNA5 no-return float atomics (resolved at L2 atomic units).
__device__ __forceinline__ void gatomic_add(float* p, float v) {
  asm volatile("global_atomic_add_f32 %0, %1, off"
               :: "v"((unsigned long long)(size_t)p), "v"(v) : "memory");
}
__device__ __forceinline__ void gatomic_max(float* p, float v) {
  asm volatile("global_atomic_max_num_f32 %0, %1, off"
               :: "v"((unsigned long long)(size_t)p), "v"(v) : "memory");
}
__device__ __forceinline__ void gatomic_min(float* p, float v) {
  asm volatile("global_atomic_min_num_f32 %0, %1, off"
               :: "v"((unsigned long long)(size_t)p), "v"(v) : "memory");
}

// ---------- small utility kernels ----------
__global__ void zero_f32(float* p, int n) {
  int i = blockIdx.x * 256 + threadIdx.x;
  if (i < n) p[i] = 0.0f;
}

__global__ void init_agg(float* s, float* q, float* mn, float* mx, int n) {
  int i = blockIdx.x * 256 + threadIdx.x;
  if (i < n) { s[i] = 0.0f; q[i] = 0.0f; mn[i] = 3.402823466e38f; mx[i] = -3.402823466e38f; }
}

__global__ void deg_count(const int* __restrict__ dstI, float* deg, int E) {
  int i = blockIdx.x * 256 + threadIdx.x;
  if (i < E) gatomic_add(deg + dstI[i], 1.0f);
}

__global__ void cvt_bf16(const float* __restrict__ in, unsigned short* __restrict__ out, int n) {
  int i = blockIdx.x * 256 + threadIdx.x;
  if (i < n) out[i] = (unsigned short)f2bf(in[i]);
}

// ---------- edge kernel: message GEMM + 4-way atomic scatter ----------
// 1 wave = 16 edges; block = 4 waves. A = [16 x 256] bf16 in LDS, B = pre_w (bf16, [128][256]).
#define WPB_E 4
#define ROWB_E 528   // 256 bf16 + 16B pad, multiple of 16

__global__ __launch_bounds__(128) void edge_kernel(
    const float* __restrict__ x, const int* __restrict__ srcI, const int* __restrict__ dstI,
    const unsigned short* __restrict__ preW, const float* __restrict__ preB,
    float* aggS, float* aggQ, float* aggMn, float* aggMx, int nTiles, int E)
{
  alignas(16) __shared__ unsigned char sm[WPB_E * 16 * ROWB_E];
  const int lane = threadIdx.x & 31;
  const int wv   = threadIdx.x >> 5;
  const int tile = blockIdx.x * WPB_E + wv;
  const bool active = (tile < nTiles);
  unsigned char* myA = sm + wv * (16 * ROWB_E);
  const int erow = lane & 15;

  if (active) {
    int e = tile * 16 + erow; if (e >= E) e = E - 1;
    int half = lane >> 4;                       // 0: x[dst] block, 1: x[src] block
    int row  = (half == 0) ? dstI[e] : srcI[e];
    const float* xr = x + (size_t)row * FDIM;
    unsigned char* dp = myA + erow * ROWB_E + half * 256;  // 128 bf16 per half
#pragma unroll
    for (int c = 0; c < FDIM; c += 4) {
      f32x4 v = *(const f32x4*)(xr + c);
      u32x2 pk; pk.x = pk_bf16(v.x, v.y); pk.y = pk_bf16(v.z, v.w);
      *(u32x2*)(dp + c * 2) = pk;
    }
  }
  __syncthreads();
  if (!active) return;

  v8f acc[8] = {};
#pragma unroll
  for (int kk = 0; kk < 8; ++kk) {
    Frag aF;
    const unsigned char* ap = myA + (lane & 15) * ROWB_E + kk * 64 + ((lane < 16) ? 0 : 16);
    aF.q[0] = *(const u32x4*)ap;
    aF.q[1] = *(const u32x4*)(ap + 32);
#pragma unroll
    for (int ct = 0; ct < 8; ++ct) {
      Frag bF;
      const unsigned char* bp = (const unsigned char*)preW
          + (size_t)(ct * 16 + (lane & 15)) * 512 + kk * 64 + ((lane < 16) ? 0 : 16);
      bF.q[0] = *(const u32x4*)bp;
      bF.q[1] = *(const u32x4*)(bp + 32);
      acc[ct] = wmma_bf16(aF.v, bF.v, acc[ct]);
    }
  }

#pragma unroll
  for (int ct = 0; ct < 8; ++ct) {
    int feat = ct * 16 + (lane & 15);
    float bv = preB[feat];
#pragma unroll
    for (int r = 0; r < 8; ++r) {
      int M = (lane < 16) ? r : (r + 8);
      int e = tile * 16 + M;
      if (e < E) {
        int d = dstI[e];
        float v = acc[ct][r] + bv;
        size_t o = (size_t)d * FDIM + feat;
        gatomic_add(aggS  + o, v);
        gatomic_add(aggQ  + o, v * v);
        gatomic_max(aggMx + o, v);
        gatomic_min(aggMn + o, v);
      }
    }
  }
}

// ---------- node kernel: build 13F concat, post GEMM, lin GEMM, ReLU ----------
// 1 block = 16 nodes, 8 waves; wave w owns output column tile w.
#define ROWB_A 3344  // 1664 bf16 + 16B pad
#define ROWB_O 272   // 128 bf16 + 16B pad

__global__ __launch_bounds__(256) void node_kernel(
    const float* __restrict__ x, const float* __restrict__ deg,
    const float* __restrict__ aggS, const float* __restrict__ aggQ,
    const float* __restrict__ aggMn, const float* __restrict__ aggMx,
    const unsigned short* __restrict__ postW, const float* __restrict__ postB,
    const unsigned short* __restrict__ linW, const float* __restrict__ linB,
    const float* __restrict__ adlp, float* __restrict__ out, int Ntot)
{
  alignas(16) __shared__ unsigned char smA[16 * ROWB_A];
  alignas(16) __shared__ unsigned char smO[16 * ROWB_O];
  const int t = threadIdx.x;
  const int lane = t & 31, wv = t >> 5;
  const int nb = blockIdx.x * 16;
  const float adl = adlp[0];

  // Phase 1: cooperative build of [16 x 1664] bf16 concat rows.
#pragma unroll
  for (int it = 0; it < 8; ++it) {
    int idx = it * 256 + t;
    int r = idx >> 7, f = idx & 127;
    int n = nb + r; if (n >= Ntot) n = Ntot - 1;
    size_t o = (size_t)n * FDIM + f;
    float dg = deg[n];
    float inv = 1.0f / fmaxf(dg, 1.0f);
    float mean  = aggS[o] * inv;
    float mean2 = aggQ[o] * inv;
    float sd = sqrtf(fmaxf(mean2 - mean * mean, 0.0f) + 1e-5f);
    bool has = dg > 0.0f;
    float mn = has ? aggMn[o] : 0.0f;
    float mx = has ? aggMx[o] : 0.0f;
    float sc  = logf(fmaxf(dg, 1.0f) + 1.0f) / adl;
    float isc = 1.0f / sc;
    unsigned short* rp = (unsigned short*)(smA + r * ROWB_A);
    rp[f]          = (unsigned short)f2bf(x[o]);
    rp[128  + f]   = (unsigned short)f2bf(mean);
    rp[256  + f]   = (unsigned short)f2bf(mn);
    rp[384  + f]   = (unsigned short)f2bf(mx);
    rp[512  + f]   = (unsigned short)f2bf(sd);
    rp[640  + f]   = (unsigned short)f2bf(mean * sc);
    rp[768  + f]   = (unsigned short)f2bf(mn * sc);
    rp[896  + f]   = (unsigned short)f2bf(mx * sc);
    rp[1024 + f]   = (unsigned short)f2bf(sd * sc);
    rp[1152 + f]   = (unsigned short)f2bf(mean * isc);
    rp[1280 + f]   = (unsigned short)f2bf(mn * isc);
    rp[1408 + f]   = (unsigned short)f2bf(mx * isc);
    rp[1536 + f]   = (unsigned short)f2bf(sd * isc);
  }
  __syncthreads();

  // Phase 2: post GEMM, K = 1664 (52 WMMA steps).
  const int feat = wv * 16 + (lane & 15);
  v8f acc = {};
#pragma unroll 4
  for (int kk = 0; kk < 52; ++kk) {
    Frag aF, bF;
    const unsigned char* ap = smA + (lane & 15) * ROWB_A + kk * 64 + ((lane < 16) ? 0 : 16);
    aF.q[0] = *(const u32x4*)ap;
    aF.q[1] = *(const u32x4*)(ap + 32);
    const unsigned char* bp = (const unsigned char*)postW
        + (size_t)feat * 3328 + kk * 64 + ((lane < 16) ? 0 : 16);
    bF.q[0] = *(const u32x4*)bp;
    bF.q[1] = *(const u32x4*)(bp + 32);
    acc = wmma_bf16(aF.v, bF.v, acc);
  }
  {
    float pb = postB[feat];
#pragma unroll
    for (int r = 0; r < 8; ++r) {
      int M = (lane < 16) ? r : (r + 8);
      unsigned short* p = (unsigned short*)(smO + M * ROWB_O);
      p[feat] = (unsigned short)f2bf(acc[r] + pb);
    }
  }
  __syncthreads();

  // Phase 3: lin GEMM, K = 128 (4 WMMA steps), bias + ReLU, store.
  v8f acc2 = {};
#pragma unroll
  for (int kk = 0; kk < 4; ++kk) {
    Frag aF, bF;
    const unsigned char* ap = smO + (lane & 15) * ROWB_O + kk * 64 + ((lane < 16) ? 0 : 16);
    aF.q[0] = *(const u32x4*)ap;
    aF.q[1] = *(const u32x4*)(ap + 32);
    const unsigned char* bp = (const unsigned char*)linW
        + (size_t)feat * 256 + kk * 64 + ((lane < 16) ? 0 : 16);
    bF.q[0] = *(const u32x4*)bp;
    bF.q[1] = *(const u32x4*)(bp + 32);
    acc2 = wmma_bf16(aF.v, bF.v, acc2);
  }
  float lb = linB[feat];
#pragma unroll
  for (int r = 0; r < 8; ++r) {
    int M = (lane < 16) ? r : (r + 8);
    int n = nb + M;
    if (n < Ntot) out[(size_t)n * FDIM + feat] = fmaxf(acc2[r] + lb, 0.0f);
  }
}

// ---------- host ----------
extern "C" void kernel_launch(void* const* d_in, const int* in_sizes, int n_in,
                              void* d_out, int out_size, void* d_ws, size_t ws_size,
                              hipStream_t stream) {
  const int N = in_sizes[0] / FDIM;
  const int E = in_sizes[1] / 2;
  const float* x0  = (const float*)d_in[0];
  const int*   ei  = (const int*)d_in[1];
  const int*   srcI = ei;
  const int*   dstI = ei + E;

  char* ws = (char*)d_ws;
  size_t off = 0;
  auto take = [&](size_t bytes) -> char* {
    char* p = ws + off;
    off += (bytes + 255) & ~(size_t)255;
    return p;
  };
  const size_t NF = (size_t)N * FDIM;
  float* deg   = (float*)take((size_t)N * 4);
  float* hA    = (float*)take(NF * 4);
  float* hB    = (float*)take(NF * 4);
  float* aggS  = (float*)take(NF * 4);
  float* aggQ  = (float*)take(NF * 4);
  float* aggMn = (float*)take(NF * 4);
  float* aggMx = (float*)take(NF * 4);
  unsigned short* preW[5];  unsigned short* postW[5];  unsigned short* linW[5];
  for (int i = 0; i < 5; ++i) {
    preW[i]  = (unsigned short*)take((size_t)FDIM * 2 * FDIM * 2);   // 128x256 bf16
    postW[i] = (unsigned short*)take((size_t)FDIM * 13 * FDIM * 2);  // 128x1664 bf16
    linW[i]  = (unsigned short*)take((size_t)FDIM * FDIM * 2);       // 128x128 bf16
  }

  // Degrees (rebuilt every call for determinism).
  zero_f32<<<(N + 255) / 256, 256, 0, stream>>>(deg, N);
  deg_count<<<(E + 255) / 256, 256, 0, stream>>>(dstI, deg, E);

  // Repack weights to bf16.
  for (int i = 0; i < 5; ++i) {
    const float* pw = (const float*)d_in[2 + 7 * i + 0];
    const float* ow = (const float*)d_in[2 + 7 * i + 3];
    const float* lw = (const float*)d_in[2 + 7 * i + 5];
    int npre = FDIM * 2 * FDIM, npost = FDIM * 13 * FDIM, nlin = FDIM * FDIM;
    cvt_bf16<<<(npre  + 255) / 256, 256, 0, stream>>>(pw, preW[i],  npre);
    cvt_bf16<<<(npost + 255) / 256, 256, 0, stream>>>(ow, postW[i], npost);
    cvt_bf16<<<(nlin  + 255) / 256, 256, 0, stream>>>(lw, linW[i],  nlin);
  }

  const float* hin = x0;
  for (int i = 0; i < 5; ++i) {
    const float* preB  = (const float*)d_in[2 + 7 * i + 1];
    const float* adl   = (const float*)d_in[2 + 7 * i + 2];
    const float* postB = (const float*)d_in[2 + 7 * i + 4];
    const float* linB  = (const float*)d_in[2 + 7 * i + 6];
    float* hout = (i == 4) ? (float*)d_out : ((i % 2 == 0) ? hA : hB);

    init_agg<<<(int)((NF + 255) / 256), 256, 0, stream>>>(aggS, aggQ, aggMn, aggMx, (int)NF);

    int nTiles = (E + 15) / 16;
    edge_kernel<<<(nTiles + WPB_E - 1) / WPB_E, 128, 0, stream>>>(
        hin, srcI, dstI, preW[i], preB, aggS, aggQ, aggMn, aggMx, nTiles, E);

    node_kernel<<<(N + 15) / 16, 256, 0, stream>>>(
        hin, deg, aggS, aggQ, aggMn, aggMx,
        postW[i], postB, linW[i], linB, adl, hout, N);

    hin = hout;
  }
}
